// MGCNNet_41549513622133
// MI455X (gfx1250) — matrix operand order repaired and checked by reading, not compiled, not run on hardware
//
#include <hip/hip_runtime.h>

// ---------------------------------------------------------------------------
// GCN (2-layer) for MI455X / gfx1250.  Dense GEMMs run on v_wmma_f32_16x16x32_bf16.
// ---------------------------------------------------------------------------

#define GCN_D 768  // hidden size

typedef __attribute__((ext_vector_type(16))) __bf16 v16bf;
typedef __attribute__((ext_vector_type(8)))  float  v8f;

union FragBF16 { v16bf v; uint4 q[2]; };
union Pack4BF  { __bf16 h[4]; uint2 u; };

// ---------------------------- degree handling ------------------------------

__global__ void deg_count_kernel(const int* __restrict__ src,
                                 const int* __restrict__ dst,
                                 float* __restrict__ outc,
                                 float* __restrict__ inc, int E) {
    int e = blockIdx.x * blockDim.x + threadIdx.x;
    if (e < E) {
        atomicAdd(&outc[src[e]], 1.0f);
        atomicAdd(&inc[dst[e]], 1.0f);
    }
}

__global__ void deg_factor_kernel(float* __restrict__ outc,
                                  float* __restrict__ inc, int n) {
    int i = blockIdx.x * blockDim.x + threadIdx.x;
    if (i < n) {
        float a = outc[i];
        float b = inc[i];
        outc[i] = rsqrtf(a < 1.0f ? 1.0f : a);
        inc[i]  = rsqrtf(b < 1.0f ? 1.0f : b);
    }
}

// ------------------------- row scaling (fp32) ------------------------------
// y[i,:] = x[i,:] * f[i]   (float4 vectorized; n4 = N*D/4)

__global__ void scale_rows_kernel(const float* __restrict__ x,
                                  const float* __restrict__ f,
                                  float* __restrict__ y, int n4) {
    int i = blockIdx.x * blockDim.x + threadIdx.x;
    if (i < n4) {
        int row = (i * 4) / GCN_D;
        float  s = f[row];
        float4 v = ((const float4*)x)[i];
        v.x *= s; v.y *= s; v.z *= s; v.w *= s;
        ((float4*)y)[i] = v;
    }
}

// --------------------- SpMM scatter-add (fp32 atomics) ---------------------
// one edge per block; 192 threads * float4 = 768 elements

__global__ void scatter_add_kernel(const int* __restrict__ src,
                                   const int* __restrict__ dst,
                                   const float* __restrict__ h,
                                   float* __restrict__ agg) {
    int e = blockIdx.x;
    int s = src[e];
    int d = dst[e];
    const float4* hp = (const float4*)(h + (size_t)s * GCN_D);
    float* ap = agg + (size_t)d * GCN_D;
    int j = threadIdx.x;                 // 0..191
    float4 v = hp[j];
    int base = j * 4;
    atomicAdd(&ap[base + 0], v.x);
    atomicAdd(&ap[base + 1], v.y);
    atomicAdd(&ap[base + 2], v.z);
    atomicAdd(&ap[base + 3], v.w);
}

// ------------------ scale + fp32 -> bf16 conversion ------------------------
// xb[i,:] = bf16( agg[i,:] * f[i] )

__global__ void scale_cvt_kernel(const float* __restrict__ agg,
                                 const float* __restrict__ f,
                                 __bf16* __restrict__ xb, int n4) {
    int i = blockIdx.x * blockDim.x + threadIdx.x;
    if (i < n4) {
        int row = (i * 4) / GCN_D;
        float  s = f[row];
        float4 v = ((const float4*)agg)[i];
        Pack4BF p;
        p.h[0] = (__bf16)(v.x * s);
        p.h[1] = (__bf16)(v.y * s);
        p.h[2] = (__bf16)(v.z * s);
        p.h[3] = (__bf16)(v.w * s);
        ((uint2*)xb)[i] = p.u;
    }
}

__global__ void cvt_bf16_kernel(const float* __restrict__ w,
                                __bf16* __restrict__ wb, int n4) {
    int i = blockIdx.x * blockDim.x + threadIdx.x;
    if (i < n4) {
        float4 v = ((const float4*)w)[i];
        Pack4BF p;
        p.h[0] = (__bf16)v.x;
        p.h[1] = (__bf16)v.y;
        p.h[2] = (__bf16)v.z;
        p.h[3] = (__bf16)v.w;
        ((uint2*)wb)[i] = p.u;
    }
}

// ----------------------------- WMMA GEMM -----------------------------------
// C[M x 768] = X[M x 768](bf16) @ W[768 x 768](bf16, row-major) + bias
//              (+ res1 + res2 when provided)
// Block: 256 threads = 8 waves, 2(M) x 4(N); wave tile 64x32; block tile 128x128.
// A fragment: lane (l&15) = row, chunks at K = k0+(l>>4)*8 and +16.  (ISA 7.12.2)
// B fragment: lane (l&15) = col of W, same K chunks as A (strided).       (ISA 7.12.5)

__global__ __launch_bounds__(256)
void gemm_wmma_bf16_kernel(const __bf16* __restrict__ X,
                           const __bf16* __restrict__ W,
                           const float*  __restrict__ bias,
                           const float*  __restrict__ res1,
                           const float*  __restrict__ res2,
                           float* __restrict__ C, int M) {
    const int lane  = threadIdx.x & 31;
    const int wave  = threadIdx.x >> 5;
    const int waveM = wave & 1;   // 0..1
    const int waveN = wave >> 1;  // 0..3
    const int lr = lane & 15;     // sub-row / sub-col
    const int kg = lane >> 4;     // K-group 0/1

    const int mBase = blockIdx.y * 128 + waveM * 64;
    const int nBase = blockIdx.x * 128 + waveN * 32;

    v8f acc[4][2];
#pragma unroll
    for (int mi = 0; mi < 4; ++mi)
#pragma unroll
        for (int ni = 0; ni < 2; ++ni)
            acc[mi][ni] = (v8f){0.f, 0.f, 0.f, 0.f, 0.f, 0.f, 0.f, 0.f};

    // clamp A rows so the last (partial) M-block reads stay in bounds
    int rowA[4];
#pragma unroll
    for (int mi = 0; mi < 4; ++mi) {
        int r = mBase + mi * 16 + lr;
        rowA[mi] = (r < M) ? r : (M - 1);
    }

    for (int k0 = 0; k0 < GCN_D; k0 += 32) {
        FragBF16 a[4];
#pragma unroll
        for (int mi = 0; mi < 4; ++mi) {
            const __bf16* p = X + (size_t)rowA[mi] * GCN_D + (k0 + kg * 8);
            a[mi].q[0] = *(const uint4*)(p);
            a[mi].q[1] = *(const uint4*)(p + 16);
        }
        FragBF16 b[2];
        const int kB = k0 + kg * 8;  // lane&15 <-> N col of W; K = kB+s, kB+16+s
#pragma unroll
        for (int ni = 0; ni < 2; ++ni) {
            const __bf16* p = W + (size_t)kB * GCN_D + (nBase + ni * 16 + lr);
#pragma unroll
            for (int s = 0; s < 8; ++s) {
                b[ni].v[s]     = p[(size_t)s * GCN_D];
                b[ni].v[8 + s] = p[(size_t)(16 + s) * GCN_D];
            }
        }
#pragma unroll
        for (int mi = 0; mi < 4; ++mi)
#pragma unroll
            for (int ni = 0; ni < 2; ++ni)
                acc[mi][ni] = __builtin_amdgcn_wmma_f32_16x16x32_bf16(
                    /*neg_a=*/false, a[mi].v, /*neg_b=*/false, b[ni].v,
                    /*c_mod=*/(short)0, acc[mi][ni],
                    /*reuse_a=*/false, /*reuse_b=*/false);
    }

    // epilogue: C VGPR r -> row (mBase + mi*16 + kg*8 + r), col (nBase + ni*16 + lr)
#pragma unroll
    for (int mi = 0; mi < 4; ++mi) {
        const int rowBase = mBase + mi * 16 + kg * 8;
#pragma unroll
        for (int ni = 0; ni < 2; ++ni) {
            const int col = nBase + ni * 16 + lr;
            const float bv = bias[col];
            if (res1 != nullptr) {
#pragma unroll
                for (int r = 0; r < 8; ++r) {
                    int row = rowBase + r;
                    if (row < M) {
                        size_t idx = (size_t)row * GCN_D + col;
                        C[idx] = acc[mi][ni][r] + bv + res1[idx] + res2[idx];
                    }
                }
            } else {
#pragma unroll
                for (int r = 0; r < 8; ++r) {
                    int row = rowBase + r;
                    if (row < M) {
                        size_t idx = (size_t)row * GCN_D + col;
                        C[idx] = acc[mi][ni][r] + bv;
                    }
                }
            }
        }
    }
}

// ------------------------------- launcher ----------------------------------

extern "C" void kernel_launch(void* const* d_in, const int* in_sizes, int n_in,
                              void* d_out, int out_size, void* d_ws, size_t ws_size,
                              hipStream_t stream) {
    const float* feat = (const float*)d_in[0];
    const float* W1   = (const float*)d_in[1];
    const float* b1   = (const float*)d_in[2];
    const float* W2   = (const float*)d_in[3];
    const float* b2   = (const float*)d_in[4];
    const int*   src  = (const int*)d_in[5];
    const int*   dst  = (const int*)d_in[6];

    const int D = GCN_D;
    const int N = in_sizes[0] / D;   // 50000
    const int E = in_sizes[5];       // 450000
    const size_t nd = (size_t)N * D;

    // workspace carve-out (256B aligned slices)
    char* ws = (char*)d_ws;
    size_t off = 0;
    auto carve = [&](size_t bytes) -> char* {
        char* p = ws + off;
        off += (bytes + 255) & ~(size_t)255;
        return p;
    };
    float*  hA   = (float*)carve(nd * sizeof(float));    // scaled rows / h2
    float*  hB   = (float*)carve(nd * sizeof(float));    // agg1 -> gcn1
    __bf16* Xb   = (__bf16*)carve(nd * sizeof(__bf16));  // bf16 GEMM input
    float*  outf = (float*)carve((size_t)N * sizeof(float));
    float*  inf_ = (float*)carve((size_t)N * sizeof(float));
    __bf16* W1b  = (__bf16*)carve((size_t)D * D * sizeof(__bf16));
    __bf16* W2b  = (__bf16*)carve((size_t)D * D * sizeof(__bf16));
    float*  out  = (float*)d_out;

    const int nd4 = (int)(nd / 4);
    const int dd4 = (D * D) / 4;
    const int TB = 256;

    // --- degrees -> normalization factors ---
    hipMemsetAsync(outf, 0, (size_t)N * sizeof(float), stream);
    hipMemsetAsync(inf_, 0, (size_t)N * sizeof(float), stream);
    deg_count_kernel<<<(E + TB - 1) / TB, TB, 0, stream>>>(src, dst, outf, inf_, E);
    deg_factor_kernel<<<(N + TB - 1) / TB, TB, 0, stream>>>(outf, inf_, N);

    // --- weights to bf16 (row-major; B fragments read columns with stride D) ---
    cvt_bf16_kernel<<<(dd4 + TB - 1) / TB, TB, 0, stream>>>(W1, W1b, dd4);
    cvt_bf16_kernel<<<(dd4 + TB - 1) / TB, TB, 0, stream>>>(W2, W2b, dd4);

    dim3 gemmGrid(D / 128, (N + 127) / 128);

    // --- layer 1 ---
    scale_rows_kernel<<<(nd4 + TB - 1) / TB, TB, 0, stream>>>(feat, outf, hA, nd4);
    hipMemsetAsync(hB, 0, nd * sizeof(float), stream);
    scatter_add_kernel<<<E, D / 4, 0, stream>>>(src, dst, hA, hB);
    scale_cvt_kernel<<<(nd4 + TB - 1) / TB, TB, 0, stream>>>(hB, inf_, Xb, nd4);
    // gcn1 = Xb @ W1 + b1   (writes hB; GEMM reads only Xb/W1b)
    gemm_wmma_bf16_kernel<<<gemmGrid, 256, 0, stream>>>(Xb, W1b, b1,
                                                        nullptr, nullptr, hB, N);

    // --- layer 2 (d_out doubles as agg2 scratch, then fully overwritten) ---
    scale_rows_kernel<<<(nd4 + TB - 1) / TB, TB, 0, stream>>>(hB, outf, hA, nd4);
    hipMemsetAsync(out, 0, nd * sizeof(float), stream);
    scatter_add_kernel<<<E, D / 4, 0, stream>>>(src, dst, hA, out);
    scale_cvt_kernel<<<(nd4 + TB - 1) / TB, TB, 0, stream>>>(out, inf_, Xb, nd4);
    // out = feat + gcn1 + (Xb @ W2 + b2)
    gemm_wmma_bf16_kernel<<<gemmGrid, 256, 0, stream>>>(Xb, W2b, b2,
                                                        feat, hB, out, N);
}
